// HGCNEncoder_17463337025614
// MI455X (gfx1250) — matrix operand order repaired and checked by reading, not compiled, not run on hardware
//
#include <hip/hip_runtime.h>
#include <hip/hip_bf16.h>
#include <stdint.h>

// ---------------------------------------------------------------------------
// HGCN encoder for MI455X (gfx1250): 3 x [ WMMA bf16 GEMM -> scatter ->
// scale -> gather -> scale+bias+relu+BN-stats -> BN apply ]
// ---------------------------------------------------------------------------

typedef __attribute__((ext_vector_type(16))) __bf16 v16bf;
typedef __attribute__((ext_vector_type(8)))  float  v8f;

#define D 128
#define SXP 68               // padded LDS row stride (dwords) for packed-bf16 X tile
#define GEMM_ROWS 128        // rows per workgroup in GEMM

__device__ __forceinline__ void atomf(float* p, float v) {
    __hip_atomic_fetch_add(p, v, __ATOMIC_RELAXED, __HIP_MEMORY_SCOPE_AGENT);
}

// round-to-nearest-even f32 -> bf16, packed pair (lo in [15:0], hi in [31:16])
__device__ __forceinline__ uint32_t pkbf(float lo, float hi) {
    uint32_t a = __float_as_uint(lo);
    uint32_t b = __float_as_uint(hi);
    a = (a + 0x7FFFu + ((a >> 16) & 1u)) >> 16;
    b = (b + 0x7FFFu + ((b >> 16) & 1u)) >> 16;
    return a | (b << 16);
}

// ---------------------------------------------------------------------------
// Degree computation (once per call; src/edg are fixed across layers)
// ---------------------------------------------------------------------------
__global__ void k_deg(const long long* __restrict__ ei, float* dn, float* de, int nnz) {
    int k = blockIdx.x * 256 + threadIdx.x;
    if (k < nnz) {
        atomf(&dn[(int)ei[k]], 1.0f);
        atomf(&de[(int)ei[(size_t)nnz + k]], 1.0f);
    }
}

__global__ void k_invdeg(float* dn, int n, float* de, int e) {
    int i = blockIdx.x * 256 + threadIdx.x;
    if (i < n) { float v = dn[i]; dn[i] = v > 0.0f ? 1.0f / v : 0.0f; }
    if (i < e) { float v = de[i]; de[i] = v > 0.0f ? 1.0f / v : 0.0f; }
}

// ---------------------------------------------------------------------------
// Pack W (128x128 f32, row-major, y = x @ W) into WMMA B-fragment layout.
// B tile = 32x16 (KxN) bf16.  VGPR j, lane n (0..15): K=2j,2j+1 ;
// lane n+16: K=16+2j,17+2j.  Index: (((nt*4+kc)*32+lane)*8 + j)
// ---------------------------------------------------------------------------
__global__ void k_packW(const float* __restrict__ W, uint32_t* __restrict__ Wpk) {
    int t = blockIdx.x * 256 + threadIdx.x;
    if (t >= 8192) return;
    int j    = t & 7;
    int lane = (t >> 3) & 31;
    int kc   = (t >> 8) & 3;
    int nt   = t >> 10;
    int n    = nt * 16 + (lane & 15);
    int half = lane >> 4;
    int k0   = kc * 32 + 2 * j + 16 * half;
    Wpk[t] = pkbf(W[k0 * D + n], W[(k0 + 1) * D + n]);
}

// ---------------------------------------------------------------------------
// GEMM: Y[N x 128] = X[N x 128] @ W (via packed Wpk), bf16 WMMA, f32 accum.
// 256 threads = 8 waves; each wave owns a 16-row slab, all 8 column tiles.
// X is converted to packed bf16 once at LDS staging; the A fragment then
// loads as two contiguous ds_load_b128 per k-chunk (no VALU in inner loop).
// ---------------------------------------------------------------------------
__global__ __launch_bounds__(256) void k_gemm(const float* __restrict__ X,
                                              const uint32_t* __restrict__ Wpk,
                                              float* __restrict__ Y, int nrows) {
    __shared__ uint32_t sW[8192];              // 32 KB packed W fragments
    __shared__ uint32_t sXp[GEMM_ROWS * SXP];  // ~34 KB packed-bf16 X tile

    const int t    = threadIdx.x;
    const int wave = t >> 5;
    const int lane = t & 31;
    const int rowBase = blockIdx.x * GEMM_ROWS;

    for (int i = t; i < 8192; i += 256) sW[i] = Wpk[i];

    // stage 128x128 X tile, converting f32 -> packed bf16 pairs (zero-pad tail)
    for (int i = t; i < GEMM_ROWS * 32; i += 256) {
        int r  = i >> 5;
        int c4 = i & 31;                 // covers cols 4*c4 .. 4*c4+3
        float4 v = make_float4(0.f, 0.f, 0.f, 0.f);
        int gr = rowBase + r;
        if (gr < nrows) v = ((const float4*)X)[(size_t)gr * 32 + c4];
        uint2 p;
        p.x = pkbf(v.x, v.y);
        p.y = pkbf(v.z, v.w);
        *(uint2*)&sXp[r * SXP + 2 * c4] = p;
    }
    __syncthreads();

    const int mBase = wave * 16;
    const int mrow  = mBase + (lane & 15);
    const int half  = lane >> 4;
    const uint32_t* arow = &sXp[mrow * SXP];

    v8f c[8];
    v8f vz = {0.f, 0.f, 0.f, 0.f, 0.f, 0.f, 0.f, 0.f};
#pragma unroll
    for (int i = 0; i < 8; ++i) c[i] = vz;

#pragma unroll
    for (int kc = 0; kc < 4; ++kc) {
        // A fragment 16x32 bf16: lane m(0..15) dword j -> K=2j+16h (contiguous, matches B)
        // packed-pair dwords: q0 = arow[kc*16 + 8h + 0..3], q1 = arow[kc*16 + 8h + 4..7]
        union { uint4 q[2]; v16bf v; } A;
        A.q[0] = *(const uint4*)&arow[kc * 16 + 8 * half];
        A.q[1] = *(const uint4*)&arow[kc * 16 + 8 * half + 4];
#pragma unroll
        for (int nt = 0; nt < 8; ++nt) {
            union { uint4 q[2]; v16bf v; } B;
            const uint4* bp = (const uint4*)&sW[((nt * 4 + kc) * 32 + lane) << 3];
            B.q[0] = bp[0];
            B.q[1] = bp[1];
            c[nt] = __builtin_amdgcn_wmma_f32_16x16x32_bf16(
                false, A.v, false, B.v, (short)0, c[nt], false, false);
        }
    }

    // C/D layout: VGPR r, lanes 0-15: M=r ; lanes 16-31: M=r+8 ; N = lane%16
#pragma unroll
    for (int nt = 0; nt < 8; ++nt) {
        int col = nt * 16 + (lane & 15);
#pragma unroll
        for (int r = 0; r < 8; ++r) {
            int grow = rowBase + mBase + half * 8 + r;
            if (grow < nrows) Y[(size_t)grow * D + col] = c[nt][r];
        }
    }
}

// ---------------------------------------------------------------------------
// node -> hyperedge scatter: e_feat[edg[k]] += y[src[k]]   (L2-resident atomics)
// 8 incidences per 256-thread block, 32 lanes (float4) per row
// ---------------------------------------------------------------------------
#define PRE_DIST 512

__global__ void k_scatter(const long long* __restrict__ ei,
                          const float* __restrict__ y,
                          float* __restrict__ ef, int nnz) {
    int t = threadIdx.x;
    long long k = (long long)blockIdx.x * 8 + (t >> 5);
    int c4 = t & 31;
    if (k >= nnz) return;
    if (c4 == 0) {
        long long kp = k + PRE_DIST;
        if (kp < nnz) __builtin_prefetch(&y[(size_t)((int)ei[kp]) * D], 0, 1);
    }
    int s = (int)ei[k];
    int d = (int)ei[(size_t)nnz + k];
    float4 v = ((const float4*)y)[(size_t)s * 32 + c4];
    float* p = ef + (size_t)d * D + c4 * 4;
    atomf(p + 0, v.x); atomf(p + 1, v.y); atomf(p + 2, v.z); atomf(p + 3, v.w);
}

__global__ void k_scale_e(float* __restrict__ ef, const float* __restrict__ binv, int e) {
    int i = blockIdx.x * 256 + threadIdx.x;
    if (i < e * 32) {
        float s = binv[i >> 5];
        float4 v = ((float4*)ef)[i];
        v.x *= s; v.y *= s; v.z *= s; v.w *= s;
        ((float4*)ef)[i] = v;
    }
}

// hyperedge -> node gather: acc[src[k]] += e_feat[edg[k]]
__global__ void k_gather(const long long* __restrict__ ei,
                         const float* __restrict__ ef,
                         float* __restrict__ acc, int nnz) {
    int t = threadIdx.x;
    long long k = (long long)blockIdx.x * 8 + (t >> 5);
    int c4 = t & 31;
    if (k >= nnz) return;
    if (c4 == 0) {
        long long kp = k + PRE_DIST;
        if (kp < nnz) __builtin_prefetch(&ef[(size_t)((int)ei[(size_t)nnz + kp]) * D], 0, 1);
    }
    int s = (int)ei[k];
    int d = (int)ei[(size_t)nnz + k];
    float4 v = ((const float4*)ef)[(size_t)d * 32 + c4];
    float* p = acc + (size_t)s * D + c4 * 4;
    atomf(p + 0, v.x); atomf(p + 1, v.y); atomf(p + 2, v.z); atomf(p + 3, v.w);
}

// ---------------------------------------------------------------------------
// acc = relu(acc * Dinv[row] + bias[col]) in place, + per-column BN statistics
// ---------------------------------------------------------------------------
__global__ __launch_bounds__(256) void k_post(float* __restrict__ acc,
                                              const float* __restrict__ dinv,
                                              const float* __restrict__ bias,
                                              float* bnsum, float* bnsq, int nrows) {
    __shared__ float s0[D], s1[D];
    int t   = threadIdx.x;
    int col = t & 127;
    int rg  = t >> 7;
    int r0  = blockIdx.x * 128 + rg;
    float bcol = bias[col];
    float bs = 0.f, bq = 0.f;
    for (int i = 0; i < 64; ++i) {
        int r = r0 + 2 * i;
        if (r < nrows) {
            size_t idx = (size_t)r * D + col;
            float v = acc[idx] * dinv[r] + bcol;
            v = v > 0.f ? v : 0.f;
            acc[idx] = v;
            bs += v;
            bq += v * v;
        }
    }
    if (rg) { s0[col] = bs; s1[col] = bq; }
    __syncthreads();
    if (!rg) {
        bs += s0[col];
        bq += s1[col];
        atomf(&bnsum[col], bs);
        atomf(&bnsq[col], bq);
    }
}

__global__ void k_bnfin(const float* bnsum, const float* bnsq,
                        const float* __restrict__ g, const float* __restrict__ beta,
                        float* scale, float* shift, int n) {
    int c = threadIdx.x;
    float mu  = bnsum[c] / (float)n;
    float var = bnsq[c] / (float)n - mu * mu;
    float s   = g[c] * rsqrtf(var + 1e-5f);
    scale[c] = s;
    shift[c] = beta[c] - mu * s;
}

__global__ void k_bnapply(const float* __restrict__ acc,
                          const float* __restrict__ scale,
                          const float* __restrict__ shift,
                          float* __restrict__ out, int total4) {
    int i = blockIdx.x * 256 + threadIdx.x;
    if (i < total4) {
        int c = (i & 31) * 4;
        float4 v = ((const float4*)acc)[i];
        v.x = v.x * scale[c + 0] + shift[c + 0];
        v.y = v.y * scale[c + 1] + shift[c + 1];
        v.z = v.z * scale[c + 2] + shift[c + 2];
        v.w = v.w * scale[c + 3] + shift[c + 3];
        ((float4*)out)[i] = v;
    }
}

// ---------------------------------------------------------------------------
extern "C" void kernel_launch(void* const* d_in, const int* in_sizes, int n_in,
                              void* d_out, int out_size, void* d_ws, size_t ws_size,
                              hipStream_t stream) {
    const long long* ei = (const long long*)d_in[0];
    const float* node_attr = (const float*)d_in[1];
    const int NNZ = in_sizes[0] / 2;
    const int N   = in_sizes[1] / D;
    const int E   = in_sizes[2] / D;

    const float* Wl[3]    = {(const float*)d_in[3], (const float*)d_in[7],  (const float*)d_in[11]};
    const float* bl[3]    = {(const float*)d_in[4], (const float*)d_in[8],  (const float*)d_in[12]};
    const float* gl[3]    = {(const float*)d_in[5], (const float*)d_in[9],  (const float*)d_in[13]};
    const float* betal[3] = {(const float*)d_in[6], (const float*)d_in[10], (const float*)d_in[14]};

    // workspace layout (floats)
    float* wsA    = (float*)d_ws;                  // N*D
    float* wsB    = wsA + (size_t)N * D;           // N*D
    float* e_feat = wsB + (size_t)N * D;           // E*D
    float* deg_n  = e_feat + (size_t)E * D;        // N  (stores 1/deg after invert)
    float* deg_e  = deg_n + N;                     // E  (stores 1/|e| after invert)
    uint32_t* Wpk = (uint32_t*)(deg_e + E);        // 8192 packed bf16 pairs
    float* bnsum  = (float*)(Wpk + 8192);          // 128
    float* bnsq   = bnsum + D;                     // 128
    float* bnscl  = bnsq + D;                      // 128
    float* bnsft  = bnscl + D;                     // 128

    float* dout = (float*)d_out;

    // buffer rotation per layer: (x_in, y, acc, x_out)
    const float* xin[3] = {node_attr, wsA, wsB};
    float* ybuf[3]      = {wsA, wsB, wsA};
    float* accb[3]      = {wsB, dout, dout};
    float* xout[3]      = {wsA, wsB, dout};

    // degrees (recomputed every call for determinism)
    hipMemsetAsync(deg_n, 0, (size_t)(N + E) * sizeof(float), stream);
    k_deg<<<(NNZ + 255) / 256, 256, 0, stream>>>(ei, deg_n, deg_e, NNZ);
    int mx = N > E ? N : E;
    k_invdeg<<<(mx + 255) / 256, 256, 0, stream>>>(deg_n, N, deg_e, E);

    const int gemmGrid = (N + GEMM_ROWS - 1) / GEMM_ROWS;
    const int nnzGrid  = (NNZ + 7) / 8;

    for (int l = 0; l < 3; ++l) {
        // 1) pack W into WMMA fragment layout, then GEMM y = x @ W
        k_packW<<<32, 256, 0, stream>>>(Wl[l], Wpk);
        k_gemm<<<gemmGrid, 256, 0, stream>>>(xin[l], Wpk, ybuf[l], N);

        // 2) node -> hyperedge, scale by 1/|e|
        hipMemsetAsync(e_feat, 0, (size_t)E * D * sizeof(float), stream);
        k_scatter<<<nnzGrid, 256, 0, stream>>>(ei, ybuf[l], e_feat, NNZ);
        k_scale_e<<<(E * 32 + 255) / 256, 256, 0, stream>>>(e_feat, deg_e, E);

        // 3) hyperedge -> node
        hipMemsetAsync(accb[l], 0, (size_t)N * D * sizeof(float), stream);
        k_gather<<<nnzGrid, 256, 0, stream>>>(ei, e_feat, accb[l], NNZ);

        // 4) scale + bias + relu + BN stats, then BN
        hipMemsetAsync(bnsum, 0, 2 * D * sizeof(float), stream);
        k_post<<<(N + 127) / 128, 256, 0, stream>>>(accb[l], deg_n, bl[l], bnsum, bnsq, N);
        k_bnfin<<<1, D, 0, stream>>>(bnsum, bnsq, gl[l], betal[l], bnscl, bnsft, N);
        k_bnapply<<<(N * 32 + 255) / 256, 256, 0, stream>>>(accb[l], bnscl, bnsft, xout[l], N * 32);
    }
}
